// DiffLinearAttention_39341900431696
// MI455X (gfx1250) — matrix operand order repaired and checked by reading, not compiled, not run on hardware
//
#include <hip/hip_runtime.h>
#include <cmath>

typedef __attribute__((ext_vector_type(16))) __bf16 v16bf;
typedef __attribute__((ext_vector_type(8)))  __bf16 v8bf;
typedef __attribute__((ext_vector_type(8)))  float  v8f;

#define WMMA_BF16(A,B,C) \
  __builtin_amdgcn_wmma_f32_16x16x32_bf16(false,(A),false,(B),(short)0,(C),false,false)

namespace dla {
constexpr int   H = 16, L = 2048, D = 64;
constexpr int   QTILES = L / 16;                 // 128 query tiles of 16 rows
constexpr float LAMBDA_INIT = 0.2f, EPS = 1e-6f;
constexpr size_t NHLD  = (size_t)H * L * D;      // 2,097,152 elements
constexpr size_t OFF_Q1 = 0;
constexpr size_t OFF_K1 = NHLD;
constexpr size_t OFF_Q2 = 2 * NHLD;
constexpr size_t OFF_K2 = 3 * NHLD;
constexpr size_t OFF_VT = 4 * NHLD;              // V transposed [h][d][l]
constexpr size_t OFF_WT = 5 * NHLD;              // 4 * H * 64*64 bf16 (transposed W)
}

union ABu { v16bf v; v8bf h[2]; };

// Two contiguous 8x bf16 (16B) loads -> one 16-element WMMA operand fragment.
__device__ __forceinline__ v16bf ld2(const __bf16* p0, const __bf16* p1) {
  ABu u;
  u.h[0] = *reinterpret_cast<const v8bf*>(p0);
  u.h[1] = *reinterpret_cast<const v8bf*>(p1);
  return u.v;
}

// Async global->LDS 16B copy (CDNA5, ASYNCcnt-tracked).
__device__ __forceinline__ void async_b128(const __bf16* gsrc, const __bf16* lds_dst) {
  unsigned ldsoff = (unsigned)(uintptr_t)lds_dst;       // addr[31:0] = LDS byte offset
  unsigned long long ga = (unsigned long long)(uintptr_t)gsrc;
  asm volatile("global_load_async_to_lds_b128 %0, %1, off"
               :: "v"(ldsoff), "v"(ga) : "memory");
}

__device__ __forceinline__ void wait_async0() {
  asm volatile("s_wait_asynccnt 0x0" ::: "memory");
}

// ---------------------------------------------------------------------------
// Phase 0: convert the four [H,64,64] f32 weight matrices to bf16, transposed
// to [m][h][n][k] so GEMM B-fragments are two contiguous 8-element k-runs.
// grid (H, 4), block 256
// ---------------------------------------------------------------------------
__global__ void dla_wconv(const float* __restrict__ wq1, const float* __restrict__ wk1,
                          const float* __restrict__ wq2, const float* __restrict__ wk2,
                          __bf16* __restrict__ ws) {
  using namespace dla;
  const float* srcs[4] = { wq1, wk1, wq2, wk2 };
  const int h = blockIdx.x, m = blockIdx.y;
  const float* src = srcs[m] + (size_t)h * D * D;          // [k][n]
  __bf16* dst = ws + OFF_WT + ((size_t)m * H + h) * D * D; // [n][k]
  for (int i = threadIdx.x; i < D * D; i += blockDim.x) {
    const int k = i >> 6, n = i & 63;
    dst[n * 64 + k] = (__bf16)src[k * 64 + n];
  }
}

// ---------------------------------------------------------------------------
// Phase 1: feature maps. 4 waves per block, one 16-row tile per block.
// grid (QTILES, H), block 128
// ---------------------------------------------------------------------------
__global__ void dla_featmap(const float* __restrict__ Q, const float* __restrict__ K,
                            const float* __restrict__ V, __bf16* __restrict__ ws) {
  using namespace dla;
  __shared__ float ystage[4][16][64];   // 16 KB

  const int tid  = threadIdx.x;
  const int lane = tid & 31, wid = tid >> 5;
  const int qt = blockIdx.x, h = blockIdx.y;
  const int m = lane & 15, half = lane >> 4;

  const float* src = (wid < 2) ? Q : K;
  const int msel = (wid == 0) ? 0 : (wid == 1) ? 2 : (wid == 2) ? 1 : 3;

  const float*  xrow = src + (size_t)(h * L + qt * 16 + m) * D;
  const __bf16* wtb  = ws + OFF_WT + ((size_t)msel * H + h) * D * D;

  v8f acc[4] = { {}, {}, {}, {} };
  #pragma unroll
  for (int ks = 0; ks < 2; ++ks) {
    v16bf a;
    const int k0 = ks * 32 + half * 8;
    #pragma unroll
    for (int j = 0; j < 8; ++j) {
      a[j]     = (__bf16)xrow[k0 + j];
      a[8 + j] = (__bf16)xrow[k0 + 16 + j];
    }
    #pragma unroll
    for (int nt = 0; nt < 4; ++nt) {
      const __bf16* wrow = wtb + (size_t)(nt * 16 + m) * 64 + ks * 32 + half * 8;
      v16bf b = ld2(wrow, wrow + 16);
      acc[nt] = WMMA_BF16(a, b, acc[nt]);
    }
  }

  const bool sig = (wid & 1);
  #pragma unroll
  for (int nt = 0; nt < 4; ++nt)
    #pragma unroll
    for (int r = 0; r < 8; ++r) {
      float y = acc[nt][r];
      y = fmaxf(y, 0.0f);
      if (sig) y = __builtin_amdgcn_rcpf(1.0f + __expf(-y));  // fast sigmoid
      ystage[wid][r + half * 8][nt * 16 + m] = y;
    }
  __syncthreads();

  for (int i = tid; i < 16 * 64; i += blockDim.x) {
    const int row = i >> 6, col = i & 63;
    const int rg  = qt * 16 + row;
    const float x1 = ystage[0][row][col];
    const float s1 = ystage[1][row][col];
    const float y1 = ystage[2][row][col];
    const float s2 = ystage[3][row][col];
    const size_t off = (size_t)(h * L + rg) * D + col;
    ws[OFF_Q1 + off] = (__bf16)x1;
    ws[OFF_Q2 + off] = (__bf16)(s1 * x1);
    ws[OFF_K1 + off] = (__bf16)y1;
    ws[OFF_K2 + off] = (__bf16)(s2 * y1);
    ws[OFF_VT + ((size_t)h * D + col) * L + rg] = (__bf16)V[off];
  }
}

// ---------------------------------------------------------------------------
// Phase 2: causal differential attention.
// Block = 4 waves = 4 adjacent 16-query tiles of one head, sharing a block-level
// key loop. K1/K2/V^T 32-key blocks (12 KB) are staged into LDS with
// double-buffered global_load_async_to_lds_b128 (ASYNCcnt), consumed by WMMAs.
// grid (QTILES/4, H), block 128
// ---------------------------------------------------------------------------
__global__ void dla_attn(const __bf16* __restrict__ ws, float* __restrict__ out,
                         const float* __restrict__ lq1, const float* __restrict__ lk1,
                         const float* __restrict__ lq2, const float* __restrict__ lk2) {
  using namespace dla;
  // stage[buf][0]=K1 (32x64), [1]=K2 (32x64), [2]=V^T (64x32)  -> 3*4KB per buf
  __shared__ alignas(16) __bf16 stage[2][3][2048];      // 24 KB
  __shared__ alignas(16) __bf16 sst1[4][16 * 32];       // 4 KB, per-wave S1 relayout
  __shared__ alignas(16) __bf16 sst2[4][16 * 32];       // 4 KB

  const int tid  = threadIdx.x;
  const int lane = tid & 31, wid = tid >> 5;
  const int qtb = blockIdx.x * 4;        // first query tile of this block
  const int qt  = qtb + wid;             // this wave's query tile
  const int h   = blockIdx.y;
  const int m = lane & 15, half = lane >> 4;

  // lambda_full = tanh(relu(lam1 - lam2 + 0.2))
  float lam1 = 0.f, lam2 = 0.f;
  #pragma unroll 8
  for (int i = 0; i < D; ++i) { lam1 += lq1[i] * lk1[i]; lam2 += lq2[i] * lk2[i]; }
  float lam = tanhf(fmaxf(lam1 - lam2 + LAMBDA_INIT, 0.0f));

  const __bf16* q1row = ws + OFF_Q1 + (size_t)(h * L + qt * 16 + m) * D;
  const __bf16* q2row = ws + OFF_Q2 + (size_t)(h * L + qt * 16 + m) * D;
  const __bf16* k1b   = ws + OFF_K1 + (size_t)h * L * D;
  const __bf16* k2b   = ws + OFF_K2 + (size_t)h * L * D;
  const __bf16* vtb   = ws + OFF_VT + (size_t)h * D * L;

  // Hoisted, loop-invariant Q A-fragments (both K-steps, both maps).
  v16bf aq1[2], aq2[2];
  #pragma unroll
  for (int ks = 0; ks < 2; ++ks) {
    const int a0 = ks * 32 + half * 8;
    aq1[ks] = ld2(q1row + a0, q1row + a0 + 16);
    aq2[ks] = ld2(q2row + a0, q2row + a0 + 16);
  }

  v8f o1[4] = { {}, {}, {}, {} };
  v8f o2[4] = { {}, {}, {}, {} };
  float rs1[8] = {0.f}, rs2[8] = {0.f};

  const int ktwave = (qt * 16 + 15) >> 5;          // this wave's causal bound
  const int ktblk  = ((qtb + 3) * 16 + 15) >> 5;   // block-level bound (wave 3)

  // Stage one 32-key block (K1, K2, V^T) into stage[buf] cooperatively.
  auto stage_block = [&](int kt, int buf) {
    const __bf16* k1src = k1b + (size_t)kt * 32 * D;   // contiguous 4 KB
    const __bf16* k2src = k2b + (size_t)kt * 32 * D;   // contiguous 4 KB
    #pragma unroll
    for (int j = 0; j < 2; ++j) {
      const int c = tid + 128 * j;                      // chunk 0..255 (16B chunks)
      async_b128(k1src + c * 8, &stage[buf][0][c * 8]);
      async_b128(k2src + c * 8, &stage[buf][1][c * 8]);
      const int d = c >> 2, part = c & 3;               // V^T: 64 dims x 64B rows
      async_b128(vtb + (size_t)d * L + kt * 32 + part * 8, &stage[buf][2][c * 8]);
    }
  };

  stage_block(0, 0);

  for (int kt = 0; kt <= ktblk; ++kt) {
    const int buf = kt & 1;
    wait_async0();        // stage[buf] landed
    __syncthreads();      // every wave past previous consume; safe to refill buf^1
    if (kt + 1 <= ktblk) stage_block(kt + 1, buf ^ 1);

    if (kt <= ktwave) {
      const bool needmask = (kt * 32 + 31) > (qt * 16);
      const __bf16* k1l = &stage[buf][0][0];
      const __bf16* k2l = &stage[buf][1][0];
      const __bf16* vtl = &stage[buf][2][0];

      // ---- scores: two 16-key subtiles, K=64 over features ----
      #pragma unroll
      for (int s = 0; s < 2; ++s) {
        v8f c1 = {}, c2 = {};
        const __bf16* krow1 = k1l + (s * 16 + m) * D;   // B: col n = lane&15 = key
        const __bf16* krow2 = k2l + (s * 16 + m) * D;
        #pragma unroll
        for (int ks = 0; ks < 2; ++ks) {
          const int b0 = ks * 32 + half * 8;
          v16bf b1 = ld2(krow1 + b0, krow1 + b0 + 16);
          v16bf b2 = ld2(krow2 + b0, krow2 + b0 + 16);
          c1 = WMMA_BF16(aq1[ks], b1, c1);
          c2 = WMMA_BF16(aq2[ks], b2, c2);
        }
        // causal mask (diagonal blocks only) + rowsums + stage S (C layout)
        #pragma unroll
        for (int r = 0; r < 8; ++r) {
          float x1 = c1[r], x2 = c2[r];
          if (needmask) {
            const int row = qt * 16 + r + half * 8;
            const int key = kt * 32 + s * 16 + m;
            if (key > row) { x1 = 0.f; x2 = 0.f; }
          }
          rs1[r] += x1; rs2[r] += x2;
          const int sidx = (r + half * 8) * 32 + s * 16 + m;
          sst1[wid][sidx] = (__bf16)x1;
          sst2[wid][sidx] = (__bf16)x2;
        }
      }
      asm volatile("s_wait_dscnt 0x0" ::: "memory");

      // ---- PV: A = staged S (16x32), B = V^T key-runs from LDS, K=32 ----
      const __bf16* s1r = &sst1[wid][m * 32 + half * 8];
      const __bf16* s2r = &sst2[wid][m * 32 + half * 8];
      v16bf A1 = ld2(s1r, s1r + 16);
      v16bf A2 = ld2(s2r, s2r + 16);
      #pragma unroll
      for (int nt = 0; nt < 4; ++nt) {
        const __bf16* vcol = vtl + (nt * 16 + m) * 32 + half * 8;
        v16bf bv = ld2(vcol, vcol + 16);
        o1[nt] = WMMA_BF16(A1, bv, o1[nt]);
        o2[nt] = WMMA_BF16(A2, bv, o2[nt]);
      }
    }
  }

  // reduce rowsums across the 16-lane half-groups (wave32), then reciprocal
  float iv1[8], iv2[8];
  #pragma unroll
  for (int r = 0; r < 8; ++r) {
    float a = rs1[r], b = rs2[r];
    #pragma unroll
    for (int off = 1; off < 16; off <<= 1) {
      a += __shfl_xor(a, off, 32);
      b += __shfl_xor(b, off, 32);
    }
    iv1[r] = __builtin_amdgcn_rcpf(a + EPS);          // v_rcp_f32, no div chains
    iv2[r] = lam * __builtin_amdgcn_rcpf(b + EPS);
  }

  // out = O1/rs1 - lambda * O2/rs2   (C layout -> [h][row][col])
  #pragma unroll
  for (int nt = 0; nt < 4; ++nt)
    #pragma unroll
    for (int r = 0; r < 8; ++r) {
      const int row = qt * 16 + r + half * 8;
      const int col = nt * 16 + m;
      out[((size_t)h * L + row) * D + col] =
          o1[nt][r] * iv1[r] - o2[nt][r] * iv2[r];
    }
}

// ---------------------------------------------------------------------------
extern "C" void kernel_launch(void* const* d_in, const int* in_sizes, int n_in,
                              void* d_out, int out_size, void* d_ws, size_t ws_size,
                              hipStream_t stream) {
  using namespace dla;
  const float* Q   = (const float*)d_in[0];
  const float* K   = (const float*)d_in[1];
  const float* V   = (const float*)d_in[2];
  const float* Wq1 = (const float*)d_in[3];
  const float* Wk1 = (const float*)d_in[4];
  const float* Wq2 = (const float*)d_in[5];
  const float* Wk2 = (const float*)d_in[6];
  const float* lq1 = (const float*)d_in[7];
  const float* lk1 = (const float*)d_in[8];
  const float* lq2 = (const float*)d_in[9];
  const float* lk2 = (const float*)d_in[10];
  __bf16* ws  = (__bf16*)d_ws;
  float*  out = (float*)d_out;

  dla_wconv<<<dim3(H, 4), 256, 0, stream>>>(Wq1, Wk1, Wq2, Wk2, ws);
  dla_featmap<<<dim3(QTILES, H), 128, 0, stream>>>(Q, K, V, ws);
  dla_attn<<<dim3(QTILES / 4, H), 128, 0, stream>>>(ws, out, lq1, lk1, lq2, lk2);
}
